// CustomDynamicEdgeConv_49495203119849
// MI455X (gfx1250) — matrix operand order repaired and checked by reading, not compiled, not run on hardware
//
#include <hip/hip_runtime.h>

typedef unsigned short u16;
typedef __attribute__((ext_vector_type(16))) __bf16 v16bf;
typedef __attribute__((ext_vector_type(8)))  float  v8f;
typedef __attribute__((ext_vector_type(4)))  unsigned v4u;
typedef __attribute__((ext_vector_type(8)))  int      v8i;
typedef __attribute__((ext_vector_type(4)))  int      v4i;

#define D_FEAT  256
#define K1      512     // 2*D_FEAT
#define HIDDEN  512
#define D_OUT   256
#define M_TILE  64
#define THREADS 256
#define SLAB_BASE   (M_TILE * K1)     // u16 elements
#define SLAB_STRIDE (HIDDEN * 32)     // u16 elements (32 KB per slab)

#if defined(__gfx1250__) && __has_builtin(__builtin_amdgcn_tensor_load_to_lds)
#define USE_TDM 1
#else
#define USE_TDM 0
#endif

__device__ __forceinline__ u16 f2bf_sw(float f) {
  union { float f; unsigned u; } x; x.f = f;
  unsigned r = x.u + 0x7FFFu + ((x.u >> 16) & 1u);   // round-to-nearest-even
  return (u16)(r >> 16);
}
__device__ __forceinline__ unsigned pack2(float a, float b) {
#if defined(__gfx1250__) && __has_builtin(__builtin_amdgcn_cvt_pk_bf16_f32)
  auto t = __builtin_amdgcn_cvt_pk_bf16_f32(a, b);   // v_cvt_pk_bf16_f32
  union { decltype(t) v; unsigned u; } r; r.v = t;
  return r.u;
#else
  return (unsigned)f2bf_sw(a) | ((unsigned)f2bf_sw(b) << 16);
#endif
}
__device__ __forceinline__ u16 f2bf(float f) { return (u16)(pack2(f, 0.f) & 0xFFFFu); }

union Frag { uint4 u[2]; v16bf v; };

// A fragment (16x32 bf16, M=lane%16). lanes 0-15: K {0..7,16..23}; lanes 16-31: K {8..15,24..31}
__device__ __forceinline__ v16bf load_a_frag(const u16* rowk, int lane) {
  int off0 = (lane & 16) ? 8 : 0;
  Frag r;
  r.u[0] = *(const uint4*)(rowk + off0);
  r.u[1] = *(const uint4*)(rowk + off0 + 16);
  return r.v;
}
// B fragment (32x16 bf16, N=lane%16). lanes 0-15: K {0..7,16..23}; lanes 16-31: K {8..15,24..31}
__device__ __forceinline__ v16bf load_b_frag(const u16* colk, int lane) {
  int off = (lane & 16) ? 8 : 0;
  Frag r;
  const uint4* p = (const uint4*)(colk + off);
  r.u[0] = p[0];
  r.u[1] = p[2];
  return r.v;
}

#if USE_TDM
// Issue one TDM 2D tile load: tileX=32 bf16 elems (contiguous), tileY rows,
// row stride = 512 elems. D# layout per CDNA5 ISA ch.8 (groups 0/1, 2D).
__device__ __forceinline__ void tdm_issue(unsigned lds_off, const u16* gptr, unsigned tileY) {
  unsigned long long ga = (unsigned long long)(uintptr_t)gptr;
  const unsigned td0 = 0x7FFFFFFFu, td1 = 0x7FFFFFFFu;  // huge dims: no OOB clip
  const unsigned tileX = 32u, strideX = 512u;
  v4u g0;
  g0.x = 1u;                                            // count=1 (valid user D#)
  g0.y = lds_off;                                       // lds_addr (bytes)
  g0.z = (unsigned)(ga & 0xFFFFFFFFu);                  // global_addr[31:0]
  g0.w = (unsigned)((ga >> 32) & 0x01FFFFFFu) | (2u << 30);  // [56:32] | type=2
  v8i g1;
  g1[0] = (int)(1u << 16);                              // wg_mask=0, data_size=1 (2B)
  g1[1] = (int)((td0 & 0xFFFFu) << 16);                 // abar=0 | tensor_dim0.lo16
  g1[2] = (int)(((td0 >> 16) & 0xFFFFu) | ((td1 & 0xFFFFu) << 16));
  g1[3] = (int)(((td1 >> 16) & 0xFFFFu) | (tileX << 16));   // | tile_dim0
  g1[4] = (int)(tileY & 0xFFFFu);                       // tile_dim1 | tile_dim2=0
  g1[5] = (int)strideX;                                 // tensor_dim0_stride[31:0]
  g1[6] = 0;                                            // stride hi | dim1_stride lo
  g1[7] = 0;
  v4i gz = {0, 0, 0, 0};
#if __clang_major__ >= 23
  v8i gz8 = {0, 0, 0, 0, 0, 0, 0, 0};
  __builtin_amdgcn_tensor_load_to_lds(g0, g1, gz, gz, gz8, 0);
#else
  __builtin_amdgcn_tensor_load_to_lds(g0, g1, gz, gz, 0);
#endif
}
#endif

// --- prep: W1[k][n] (512x512) -> w1T[n][k] bf16 ; W2[k][n] (512x256) -> w2T[n][k] bf16 ---
__global__ void prep_w1_kernel(const float* __restrict__ W1, u16* __restrict__ w1T) {
  int i = blockIdx.x * 256 + threadIdx.x;      // 262144 elements
  int n = i & (HIDDEN - 1);
  int k = i >> 9;
  w1T[n * K1 + k] = f2bf(W1[i]);
}
__global__ void prep_w2_kernel(const float* __restrict__ W2, u16* __restrict__ w2T) {
  int i = blockIdx.x * 256 + threadIdx.x;      // 131072 elements
  int n = i & (D_OUT - 1);
  int k = i >> 8;
  w2T[n * HIDDEN + k] = f2bf(W2[i]);
}
__global__ void zero_kernel(float* __restrict__ p, int n) {
  int i = blockIdx.x * 256 + threadIdx.x;
  if (i < n) p[i] = 0.0f;
}
__global__ void deg_kernel(const int* __restrict__ nn, float* __restrict__ deg, int E) {
  int i = blockIdx.x * 256 + threadIdx.x;
  if (i < E) atomicAdd(deg + nn[E + i], 1.0f);
}
__global__ void final_kernel(float* __restrict__ out, const float* __restrict__ deg, int total) {
  int i = blockIdx.x * 256 + threadIdx.x;
  if (i < total) out[i] = out[i] / (deg[i >> 8] + 1e-8f);   // 256 cols/node
}

// --- fused per-edge MLP: feat -> relu(feat@W1+b1) -> @W2+b2 -> atomic scatter ---
__global__ void __launch_bounds__(THREADS)
edge_mlp_kernel(const float* __restrict__ x,
                const float* __restrict__ b1,
                const float* __restrict__ b2,
                const int*   __restrict__ nn,
                const u16*   __restrict__ w1T,
                const u16*   __restrict__ w2T,
                float* __restrict__ out,
                int E)
{
  extern __shared__ u16 smem[];
  u16* featA = smem;                         // 64 x 512 bf16 (64 KB); h aliases this
  u16* hA    = smem;                         // alias: feat dead before h is written

  const int tid  = threadIdx.x;
  const int lane = tid & 31;
  const int wv   = tid >> 5;    // 0..7
  const int mT   = wv & 3;      // M tile within the 64-edge block
  const int nH   = wv >> 2;     // N half
  const int edgeBase = blockIdx.x * M_TILE;

#if USE_TDM
  // LDS byte offsets of the two slabs (generic LDS address: low 32 bits = LDS offset)
  const unsigned ldsBase  = (unsigned)(uintptr_t)smem;
  const unsigned slabOff0 = ldsBase + 2u * SLAB_BASE;
  const unsigned slabOff1 = ldsBase + 2u * (SLAB_BASE + SLAB_STRIDE);
  if (tid < 32) tdm_issue(slabOff0, w1T, HIDDEN);   // prefetch k=0 slab via TDM
#endif

  // ---- Phase 0: gather x rows, build feat = [x_i, x_j - x_i] as bf16 in LDS ----
  {
    int m    = tid >> 2;        // 4 threads per edge row
    int part = tid & 3;         // each thread: 128 of 512 columns
    int e    = edgeBase + m;
    int s    = nn[e];
    int t    = nn[E + e];
    const float* xi = x + (size_t)t * D_FEAT;
    const float* xj = x + (size_t)s * D_FEAT;
    int c0   = part * 128;
    u16* dst = featA + m * K1 + c0;
    if (c0 < D_FEAT) {                       // x_i half
      #pragma unroll
      for (int j = 0; j < 16; ++j) {
        float4 f0 = *(const float4*)(xi + c0 + j * 8);
        float4 f1 = *(const float4*)(xi + c0 + j * 8 + 4);
        uint4 p;
        p.x = pack2(f0.x, f0.y); p.y = pack2(f0.z, f0.w);
        p.z = pack2(f1.x, f1.y); p.w = pack2(f1.z, f1.w);
        *(uint4*)(dst + j * 8) = p;
      }
    } else {                                 // (x_j - x_i) half, f32 subtract then round
      int cc = c0 - D_FEAT;
      #pragma unroll
      for (int j = 0; j < 16; ++j) {
        float4 a0 = *(const float4*)(xj + cc + j * 8);
        float4 a1 = *(const float4*)(xj + cc + j * 8 + 4);
        float4 c0v = *(const float4*)(xi + cc + j * 8);
        float4 c1v = *(const float4*)(xi + cc + j * 8 + 4);
        uint4 p;
        p.x = pack2(a0.x - c0v.x, a0.y - c0v.y); p.y = pack2(a0.z - c0v.z, a0.w - c0v.w);
        p.z = pack2(a1.x - c1v.x, a1.y - c1v.y); p.w = pack2(a1.z - c1v.z, a1.w - c1v.w);
        *(uint4*)(dst + j * 8) = p;
      }
    }
  }

  // ---- Phase 1: h = relu(feat @ W1 + b1), per wave: 16x16 M-tile x 256 N-cols ----
  v8f zero8 = {0.f, 0.f, 0.f, 0.f, 0.f, 0.f, 0.f, 0.f};
  v8f acc[16];
  #pragma unroll
  for (int t = 0; t < 16; ++t) acc[t] = zero8;

  const u16* aRow = featA + (mT * 16 + (lane & 15)) * K1;
  for (int i = 0; i < K1 / 32; ++i) {
    int k = i * 32;
    // slab pointer derived arithmetically from the __shared__ base so the
    // compiler keeps B-fragment loads on the DS (LDS) path, not FLAT.
#if USE_TDM
    if (tid < 32) __builtin_amdgcn_s_wait_tensorcnt(0);   // current slab DMA done
    __syncthreads();                                      // publish + prev compute done
    if (tid < 32 && i + 1 < K1 / 32)
      tdm_issue((i & 1) ? slabOff0 : slabOff1, w1T + (k + 32), HIDDEN);
    const u16* slabCur = smem + SLAB_BASE + (unsigned)(i & 1) * SLAB_STRIDE;
#else
    __syncthreads();
    #pragma unroll
    for (int r = tid; r < HIDDEN; r += THREADS) {
      const uint4* g = (const uint4*)(w1T + (size_t)r * K1 + k);
      uint4* s = (uint4*)(smem + SLAB_BASE + r * 32);
      s[0] = g[0]; s[1] = g[1]; s[2] = g[2]; s[3] = g[3];
    }
    __syncthreads();
    const u16* slabCur = smem + SLAB_BASE;
#endif
    v16bf a = load_a_frag(aRow + k, lane);
    #pragma unroll
    for (int t = 0; t < 16; ++t) {
      int n = nH * 256 + t * 16 + (lane & 15);
      v16bf b = load_b_frag(slabCur + n * 32, lane);
      acc[t] = __builtin_amdgcn_wmma_f32_16x16x32_bf16(false, a, false, b,
                                                       (short)0, acc[t], false, false);
    }
  }

  __syncthreads();            // all featA reads done -> safe to overwrite with h
#if USE_TDM
  if (tid < 32) tdm_issue(slabOff0, w2T, D_OUT);          // prefetch W2 k=0 slab
#endif

  // bias + ReLU -> h tile (bf16), overwriting featA
  #pragma unroll
  for (int t = 0; t < 16; ++t) {
    int col = nH * 256 + t * 16 + (lane & 15);
    float bias = b1[col];
    #pragma unroll
    for (int v = 0; v < 8; ++v) {
      int m = mT * 16 + ((lane & 16) ? v + 8 : v);
      float h = acc[t][v] + bias;
      hA[m * HIDDEN + col] = f2bf(h > 0.f ? h : 0.f);
    }
  }

  // ---- Phase 2: msg = h @ W2 + b2, per wave: 16x16 M-tile x 128 N-cols ----
  v8f acc2[8];
  #pragma unroll
  for (int t = 0; t < 8; ++t) acc2[t] = zero8;

  const u16* aRow2 = hA + (mT * 16 + (lane & 15)) * HIDDEN;
  for (int i = 0; i < HIDDEN / 32; ++i) {
    int k = i * 32;
#if USE_TDM
    if (tid < 32) __builtin_amdgcn_s_wait_tensorcnt(0);
    __syncthreads();                                      // slab ready + h complete
    if (tid < 32 && i + 1 < HIDDEN / 32)
      tdm_issue((i & 1) ? slabOff0 : slabOff1, w2T + (k + 32), D_OUT);
    const u16* slabCur = smem + SLAB_BASE + (unsigned)(i & 1) * SLAB_STRIDE;
#else
    __syncthreads();
    {
      int r = tid;                           // 256 rows of W2T, one per thread
      const uint4* g = (const uint4*)(w2T + (size_t)r * HIDDEN + k);
      uint4* s = (uint4*)(smem + SLAB_BASE + r * 32);
      s[0] = g[0]; s[1] = g[1]; s[2] = g[2]; s[3] = g[3];
    }
    __syncthreads();
    const u16* slabCur = smem + SLAB_BASE;
#endif
    v16bf a = load_a_frag(aRow2 + k, lane);
    #pragma unroll
    for (int t = 0; t < 8; ++t) {
      int n = nH * 128 + t * 16 + (lane & 15);
      v16bf b = load_b_frag(slabCur + n * 32, lane);
      acc2[t] = __builtin_amdgcn_wmma_f32_16x16x32_bf16(false, a, false, b,
                                                        (short)0, acc2[t], false, false);
    }
  }

  // ---- Scatter: atomic-add msg rows into out[tgt] ----
  int tg[8];
  #pragma unroll
  for (int v = 0; v < 8; ++v) {
    int m = mT * 16 + ((lane & 16) ? v + 8 : v);
    tg[v] = nn[E + edgeBase + m];
  }
  #pragma unroll
  for (int t = 0; t < 8; ++t) {
    int col = nH * 128 + t * 16 + (lane & 15);
    float bias = b2[col];
    #pragma unroll
    for (int v = 0; v < 8; ++v) {
      atomicAdd(out + (size_t)tg[v] * D_OUT + col, acc2[t][v] + bias);
    }
  }
}

extern "C" void kernel_launch(void* const* d_in, const int* in_sizes, int n_in,
                              void* d_out, int out_size, void* d_ws, size_t ws_size,
                              hipStream_t stream) {
  const float* x  = (const float*)d_in[0];
  const float* W1 = (const float*)d_in[1];
  const float* b1 = (const float*)d_in[2];
  const float* W2 = (const float*)d_in[3];
  const float* b2 = (const float*)d_in[4];
  const int*   nn = (const int*)d_in[5];
  const int E      = in_sizes[5] / 2;
  const int nNodes = in_sizes[0] / D_FEAT;

  u16*   w1T = (u16*)d_ws;                                                   // 512 KB
  u16*   w2T = (u16*)((char*)d_ws + (size_t)K1 * HIDDEN * sizeof(u16));      // 256 KB
  float* deg = (float*)((char*)d_ws + (size_t)K1 * HIDDEN * sizeof(u16)
                                    + (size_t)HIDDEN * D_OUT * sizeof(u16)); // 64 KB
  float* out = (float*)d_out;

  prep_w1_kernel<<<(K1 * HIDDEN) / 256, 256, 0, stream>>>(W1, w1T);
  prep_w2_kernel<<<(HIDDEN * D_OUT) / 256, 256, 0, stream>>>(W2, w2T);
  zero_kernel<<<(out_size + 255) / 256, 256, 0, stream>>>(out, out_size);
  zero_kernel<<<(nNodes + 255) / 256, 256, 0, stream>>>(deg, nNodes);
  deg_kernel<<<(E + 255) / 256, 256, 0, stream>>>(nn, deg, E);

  // LDS: 64 KB feat/h (aliased) + 2 x 32 KB double-buffered weight slabs = 128 KB
  size_t lds_bytes = (size_t)(M_TILE * K1 + 2 * SLAB_STRIDE) * sizeof(u16);
  edge_mlp_kernel<<<E / M_TILE, THREADS, lds_bytes, stream>>>(x, b1, b2, nn, w1T, w2T, out, E);

  final_kernel<<<(out_size + 255) / 256, 256, 0, stream>>>(out, deg, out_size);
}
